// NeuralRDE_25950192402918
// MI455X (gfx1250) — matrix operand order repaired and checked by reading, not compiled, not run on hardware
//
#include <hip/hip_runtime.h>

#define BSZ   64
#define TLEN  2049
#define CDIM  6
#define SDIM  32
#define HDIM  64
#define ODIM  16
#define WINSZ 16
#define LDIM  21
#define NWIN  128
#define VOUT  672
#define NT3   42   // 672 / 16 N-tiles in layer 3

typedef __attribute__((ext_vector_type(16))) _Float16 v16h;
typedef __attribute__((ext_vector_type(8)))  _Float16 v8h;
typedef __attribute__((ext_vector_type(8)))  float    v8f;

// softplus via raw HW transcendentals (v_exp_f32 = exp2, v_log_f32 = log2):
// softplus(x) = max(x,0) + ln2 * log2(1 + exp2(-|x| * log2e))
__device__ __forceinline__ float softplus_f(float x) {
    const float LOG2E = 1.4426950408889634f;
    const float LN2   = 0.6931471805599453f;
    float e = __builtin_amdgcn_exp2f(-fabsf(x) * LOG2E);
    return fmaxf(x, 0.0f) + LN2 * __builtin_amdgcn_logf(1.0f + e);
}

__device__ __forceinline__ v8f wmma16(v16h a, v16h b, v8f c) {
    // D = A(16x32 f16) * B(32x16 f16) + C(16x16 f32)
    return __builtin_amdgcn_wmma_f32_16x16x32_f16(
        /*neg_a=*/false, a, /*neg_b=*/false, b,
        /*c_mod=*/(short)0, c, /*reuse_a=*/false, /*reuse_b=*/false);
}

// B operand (32x16 tile of W^T) gathered from row-major W[n][k] in global:
// lane<16 holds column n, kbase = chunk ; lane>=16 kbase = chunk + 8.
// halfs 0..7 at K=kbase..+7, halfs 8..15 at K=kbase+16..+23 (matches load_a_lds).
__device__ __forceinline__ v16h load_b_global(const float* W, int ld, int n, int kbase) {
    v16h b;
    const float* p = W + (size_t)n * ld + kbase;
#pragma unroll
    for (int j = 0; j < 16; ++j) b[j] = (_Float16)p[j < 8 ? j : j + 8];
    return b;
}

// A operand (16x32 tile) gathered from row-major f16 LDS activation [m][k]:
// lane L: m = L&15, kbase = chunk + (L<16 ? 0 : 8); halfs 0..7 at K=kbase..+7,
// halfs 8..15 at K=kbase+16..+23  -> two aligned 16-byte DS loads.
__device__ __forceinline__ v16h load_a_lds(const _Float16* A, int ld, int m, int kbase) {
    const v8h lo = *reinterpret_cast<const v8h*>(A + m * ld + kbase);
    const v8h hv = *reinterpret_cast<const v8h*>(A + m * ld + kbase + 16);
    v16h a;
#pragma unroll
    for (int j = 0; j < 8; ++j) { a[j] = lo[j]; a[j + 8] = hv[j]; }
    return a;
}

// ---------------------------------------------------------------------------
// Kernel 1: depth-2 log-signature per (batch, window): disp(6) + Levy area(15)
// ---------------------------------------------------------------------------
__global__ __launch_bounds__(256) void logsig_kernel(const float* __restrict__ cv,
                                                     float* __restrict__ out) {
    int idx = blockIdx.x * 256 + threadIdx.x;
    if (idx >= BSZ * NWIN) return;
    int b = idx / NWIN, w = idx % NWIN;
    const float* p = cv + ((size_t)b * TLEN + (size_t)w * WINSZ) * CDIM;

    float prev[CDIM], disp[CDIM], a[CDIM][CDIM];
#pragma unroll
    for (int c = 0; c < CDIM; ++c) { prev[c] = p[c]; disp[c] = 0.f; }
#pragma unroll
    for (int i = 0; i < CDIM; ++i)
#pragma unroll
        for (int j = 0; j < CDIM; ++j) a[i][j] = 0.f;

    for (int s = 1; s <= WINSZ; ++s) {
        float inc[CDIM];
#pragma unroll
        for (int c = 0; c < CDIM; ++c) {
            float cur = p[(size_t)s * CDIM + c];
            inc[c] = cur - prev[c];
            prev[c] = cur;
        }
        // s_exc (exclusive prefix) == disp before this segment
#pragma unroll
        for (int i = 0; i < CDIM; ++i)
#pragma unroll
            for (int j = 0; j < CDIM; ++j) a[i][j] += disp[i] * inc[j];
#pragma unroll
        for (int c = 0; c < CDIM; ++c) disp[c] += inc[c];
    }

    float* o = out + (size_t)idx * LDIM;
#pragma unroll
    for (int c = 0; c < CDIM; ++c) o[c] = disp[c];
    int k = CDIM;
#pragma unroll
    for (int i = 0; i < CDIM; ++i)
#pragma unroll
        for (int j = i + 1; j < CDIM; ++j) o[k++] = 0.5f * (a[i][j] - a[j][i]);
}

// ---------------------------------------------------------------------------
// Kernel 2: persistent scan. One block (8 waves) per 16 batch rows.
// Weights live in registers in exact WMMA B-layout for the whole scan.
// Readout is a single WMMA per step done by wave 7 (idle during layer 1).
// ---------------------------------------------------------------------------
__global__ __launch_bounds__(256) void nrde_main(
    const float* __restrict__ cv,
    const float* __restrict__ Wi0, const float* __restrict__ bi0,
    const float* __restrict__ Wi1, const float* __restrict__ bi1,
    const float* __restrict__ Wv0, const float* __restrict__ bv0,
    const float* __restrict__ Wv1, const float* __restrict__ bv1,
    const float* __restrict__ Wv2, const float* __restrict__ bv2,
    const float* __restrict__ Wr,  const float* __restrict__ br,
    const float* __restrict__ logsig,
    float* __restrict__ out)
{
    __shared__ __align__(16) float    Y32[16 * SDIM];   // state, f32 master
    __shared__ __align__(16) float    F1[16 * SDIM];
    __shared__ __align__(16) float    F2[16 * SDIM];
    __shared__ __align__(16) _Float16 Ain[16 * SDIM];   // f16 A source for layer 1
    __shared__ __align__(16) _Float16 H1s[16 * HDIM];
    __shared__ __align__(16) _Float16 H2s[16 * HDIM];
    __shared__ __align__(16) _Float16 Vs[16 * VOUT];
    __shared__ __align__(16) float    LSs[16 * LDIM];
    __shared__ __align__(16) float    tmpH[16 * HDIM];

    const int tid  = threadIdx.x;
    const int bg   = blockIdx.x;        // batch group: rows bg*16 .. bg*16+15
    const int lane = tid & 31;
    const int wid  = tid >> 5;          // wave id 0..7
    const int hi   = lane >> 4;         // lane half
    const int ln   = lane & 15;

    // ---- register-resident weight B-tiles ----
    v16h B01[2];          // waves 0-3: Wv0 tile ; waves 4-7: Wv1 tile (2 K-chunks)
    v16h B2[6][2];        // layer-3 tiles: wave w owns N-tiles w, w+8, ..., w+40
    v16h Br;              // wave 7: readout Wr tile (32x16)
    float b2r[6];
    float b01r;
    float brr;

    if (wid < 4) {
        int n = wid * 16 + ln;
        B01[0] = load_b_global(Wv0, SDIM, n, hi ? 8 : 0);   // K = 32
        B01[1] = B01[0];                                     // unused
        b01r   = bv0[n];
    } else {
        int n = (wid - 4) * 16 + ln;
        B01[0] = load_b_global(Wv1, HDIM, n,  0 + (hi ? 8 : 0));
        B01[1] = load_b_global(Wv1, HDIM, n, 32 + (hi ? 8 : 0));
        b01r   = bv1[n];
    }
    if (wid == 7) {
        Br  = load_b_global(Wr, SDIM, ln, hi ? 8 : 0);      // out[m,o]=sum_s Y[m,s]Wr[o,s]
        brr = br[ln];
    } else {
        v16h z = {};
        Br = z; brr = 0.f;
    }
#pragma unroll
    for (int i = 0; i < 6; ++i) {
        int t3 = wid + 8 * i;
        if (t3 < NT3) {
            int n = t3 * 16 + ln;
            B2[i][0] = load_b_global(Wv2, HDIM, n,  0 + (hi ? 8 : 0));
            B2[i][1] = load_b_global(Wv2, HDIM, n, 32 + (hi ? 8 : 0));
            b2r[i]   = bv2[n];
        } else {
            v16h z = {};
            B2[i][0] = z; B2[i][1] = z; b2r[i] = 0.f;
        }
    }

    // ---- initial condition h0 = softplus(x0 Wi0^T + bi0) Wi1^T + bi1 ----
    for (int idx = tid; idx < 16 * HDIM; idx += 256) {
        int m = idx >> 6, n = idx & 63;
        float acc = bi0[n];
#pragma unroll
        for (int c2 = 0; c2 < CDIM; ++c2)
            acc += cv[((size_t)(bg * 16 + m) * TLEN) * CDIM + c2] * Wi0[n * CDIM + c2];
        tmpH[idx] = softplus_f(acc);
    }
    __syncthreads();
    for (int idx = tid; idx < 16 * SDIM; idx += 256) {
        int m = idx >> 5, s = idx & 31;
        float acc = bi1[s];
        for (int n = 0; n < HDIM; ++n) acc += tmpH[m * HDIM + n] * Wi1[s * HDIM + n];
        Y32[idx] = acc;
    }
    __syncthreads();

    // ---- vector-field evaluation: Fdst = (1/WIN) * (vf(A) . ls) ----
    auto evalVF = [&](const _Float16* Asrc, float* Fdst) {
        // layer 1 (K=32): waves 0-3, one N-tile each -> H1 (16x64)
        if (wid < 4) {
            v16h a = load_a_lds(Asrc, SDIM, ln, hi ? 8 : 0);
            v8f c;
#pragma unroll
            for (int r = 0; r < 8; ++r) c[r] = b01r;
            c = wmma16(a, B01[0], c);
            int n = wid * 16 + ln;
#pragma unroll
            for (int r = 0; r < 8; ++r)
                H1s[(r + 8 * hi) * HDIM + n] = (_Float16)softplus_f(c[r]);
        }
        __syncthreads();
        // layer 2 (K=64): waves 4-7, one N-tile each -> H2 (16x64)
        if (wid >= 4) {
            v16h a0 = load_a_lds(H1s, HDIM, ln,  0 + (hi ? 8 : 0));
            v16h a1 = load_a_lds(H1s, HDIM, ln, 32 + (hi ? 8 : 0));
            v8f c;
#pragma unroll
            for (int r = 0; r < 8; ++r) c[r] = b01r;
            c = wmma16(a0, B01[0], c);
            c = wmma16(a1, B01[1], c);
            int n = (wid - 4) * 16 + ln;
#pragma unroll
            for (int r = 0; r < 8; ++r)
                H2s[(r + 8 * hi) * HDIM + n] = (_Float16)softplus_f(c[r]);
        }
        __syncthreads();
        // layer 3 (K=64, N=672): all 8 waves, up to 6 N-tiles each -> V (16x672)
        {
            v16h a0 = load_a_lds(H2s, HDIM, ln,  0 + (hi ? 8 : 0));
            v16h a1 = load_a_lds(H2s, HDIM, ln, 32 + (hi ? 8 : 0));
#pragma unroll
            for (int i = 0; i < 6; ++i) {
                int t3 = wid + 8 * i;
                if (t3 < NT3) {                 // uniform per wave -> EXEC stays full
                    v8f c;
#pragma unroll
                    for (int r = 0; r < 8; ++r) c[r] = b2r[i];
                    c = wmma16(a0, B2[i][0], c);
                    c = wmma16(a1, B2[i][1], c);
                    int n = t3 * 16 + ln;
#pragma unroll
                    for (int r = 0; r < 8; ++r)
                        Vs[(r + 8 * hi) * VOUT + n] = (_Float16)c[r];
                }
            }
        }
        __syncthreads();
        // contract V(16,32,21) with ls(16,21), scale by 1/WIN
        for (int idx = tid; idx < 16 * SDIM; idx += 256) {
            int m = idx >> 5, s = idx & 31;
            float acc = 0.f;
#pragma unroll
            for (int l = 0; l < LDIM; ++l)
                acc += (float)Vs[m * VOUT + s * LDIM + l] * LSs[m * LDIM + l];
            Fdst[idx] = acc * (1.0f / (float)WINSZ);
        }
    };

    // ---- scan: t = 0..2048 ; readout of hidden[t] at loop top, then step ----
    for (int t = 0; t < TLEN; ++t) {
        if (t < TLEN - 1 && (t & (WINSZ - 1)) == 0) {
            int w = t >> 4;
            for (int idx = tid; idx < 16 * LDIM; idx += 256) {
                int m = idx / LDIM, l = idx - m * LDIM;
                LSs[idx] = logsig[((size_t)(bg * 16 + m) * NWIN + w) * LDIM + l];
            }
        }
        for (int idx = tid; idx < 16 * SDIM; idx += 256)
            Ain[idx] = (_Float16)Y32[idx];
        __syncthreads();

        // readout: out[:,t,:] = Y(16x32) x Wr^T(32x16) + br  -- one WMMA, wave 7
        if (wid == 7) {
            v16h a = load_a_lds(Ain, SDIM, ln, hi ? 8 : 0);
            v8f c;
#pragma unroll
            for (int r = 0; r < 8; ++r) c[r] = brr;
            c = wmma16(a, Br, c);
#pragma unroll
            for (int r = 0; r < 8; ++r)
                out[((size_t)(bg * 16 + r + 8 * hi) * TLEN + (size_t)t) * ODIM + ln] = c[r];
        }
        if (t == TLEN - 1) break;       // uniform: last hidden only needs readout

        evalVF(Ain, F1);                // f1 = vf(y) . ls / WIN
        __syncthreads();

        for (int idx = tid; idx < 16 * SDIM; idx += 256)
            Ain[idx] = (_Float16)(Y32[idx] + F1[idx]);
        __syncthreads();

        evalVF(Ain, F2);                // f2 = vf(y + f1) . ls / WIN
        __syncthreads();

        for (int idx = tid; idx < 16 * SDIM; idx += 256)
            Y32[idx] += 0.5f * (F1[idx] + F2[idx]);
        __syncthreads();
    }
}

extern "C" void kernel_launch(void* const* d_in, const int* in_sizes, int n_in,
                              void* d_out, int out_size, void* d_ws, size_t ws_size,
                              hipStream_t stream) {
    (void)in_sizes; (void)n_in; (void)out_size; (void)ws_size;
    const float* cv  = (const float*)d_in[0];
    const float* Wi0 = (const float*)d_in[1];
    const float* bi0 = (const float*)d_in[2];
    const float* Wi1 = (const float*)d_in[3];
    const float* bi1 = (const float*)d_in[4];
    const float* Wv0 = (const float*)d_in[5];
    const float* bv0 = (const float*)d_in[6];
    const float* Wv1 = (const float*)d_in[7];
    const float* bv1 = (const float*)d_in[8];
    const float* Wv2 = (const float*)d_in[9];
    const float* bv2 = (const float*)d_in[10];
    const float* Wr  = (const float*)d_in[11];
    const float* br  = (const float*)d_in[12];
    float* logsig = (float*)d_ws;          // 64*128*21 f32 = 688 KB scratch
    float* out    = (float*)d_out;

    logsig_kernel<<<(BSZ * NWIN + 255) / 256, 256, 0, stream>>>(cv, logsig);
    nrde_main<<<4, 256, 0, stream>>>(cv, Wi0, bi0, Wi1, bi1,
                                     Wv0, bv0, Wv1, bv1, Wv2, bv2,
                                     Wr, br, logsig, out);
}
